// BiochemicalDynamics_23270132810376
// MI455X (gfx1250) — matrix operand order repaired and checked by reading, not compiled, not run on hardware
//
#include <hip/hip_runtime.h>

typedef _Float16 v16h __attribute__((ext_vector_type(16)));
typedef _Float16 v8h  __attribute__((ext_vector_type(8)));
typedef float    v8f  __attribute__((ext_vector_type(8)));
typedef float    v4f  __attribute__((ext_vector_type(4)));

#define NROWS 8192
#define DIM   64
#define KSPLIT 8
#define KCHUNK (NROWS / KSPLIT)   // 1024 K per wave
#define CH     128                // K per LDS stage
#define NSTAGE (KCHUNK / CH)      // 8 stages
#define BPAD   8                  // f16 pad per LDS row (row = 272 B, 16B-aligned)
#define BROW   (CH + BPAD)        // 136 f16 per LDS row

// ---------------------------------------------------------------------------
// Kernel 1: x (fp32 [8192][64]) -> xT (f16 [64][8192]) in workspace, and zero
// the per-row diag accumulator s[8192].
// ---------------------------------------------------------------------------
__global__ void prep_kernel(const float* __restrict__ x,
                            _Float16* __restrict__ xT,
                            float* __restrict__ s) {
    int idx = blockIdx.x * blockDim.x + threadIdx.x;   // 0 .. 8192*64-1
    int i = idx >> 6;
    int n = idx & 63;
    xT[(size_t)n * NROWS + i] = (_Float16)x[idx];
    if (idx < NROWS) s[idx] = 0.0f;
}

// Raw fp32 A fragment (16 floats per lane, ISA 16-bit A-operand ordering).
struct ABuf { v4f a0, a1, a2, a3; };

__device__ inline ABuf load_a_raw(const float* p) {
    ABuf r;
    r.a0 = *(const v4f*)(p);
    r.a1 = *(const v4f*)(p + 4);
    r.a2 = *(const v4f*)(p + 16);
    r.a3 = *(const v4f*)(p + 20);
    return r;
}

__device__ inline v16h cvt_a(const ABuf& r) {
    v16h av;
#pragma unroll
    for (int j = 0; j < 4; ++j) {
        av[j]      = (_Float16)r.a0[j];
        av[4 + j]  = (_Float16)r.a1[j];
        av[8 + j]  = (_Float16)r.a2[j];
        av[12 + j] = (_Float16)r.a3[j];
    }
    return av;
}

// 2x8 f16 (B operand half-wave slice: p[0..7], p[16..23]) from an LDS row.
__device__ inline v16h load_b16(const _Float16* p) {
    v8h lo = *(const v8h*)p;
    v8h hi = *(const v8h*)(p + 16);
    v16h r;
#pragma unroll
    for (int j = 0; j < 8; ++j) { r[j] = lo[j]; r[8 + j] = hi[j]; }
    return r;
}

// Async DMA: 16 bytes global -> LDS, tracked with ASYNCcnt.
__device__ inline void async_copy16(unsigned lds_off, const void* gptr) {
    asm volatile("global_load_async_to_lds_b128 %0, %1, off"
                 :: "v"(lds_off), "v"((unsigned long long)(uintptr_t)gptr)
                 : "memory");
}

#define WMMA_F16(av, bv, c) \
    __builtin_amdgcn_wmma_f32_16x16x32_f16(false, (av), false, (bv), (short)0, (c), false, false)

// ---------------------------------------------------------------------------
// Kernel 2: block = 4 waves sharing one K-part; B chunk async-staged into
// double-buffered LDS; each wave computes a 16-row x 64-col tile of Ax with
// v_wmma_f32_16x16x32_f16, then partial s_i = x_i . (A_part x)_i.
// ---------------------------------------------------------------------------
__global__ __launch_bounds__(128)
void gemm_diag_kernel(const float* __restrict__ A,
                      const _Float16* __restrict__ xT,
                      float* __restrict__ s) {
    __shared__ __align__(16) _Float16 Bs[2][64][BROW];

    const int tid  = threadIdx.x;
    const int lane = tid & 31;
    const int wave = tid >> 5;

    const int kpart    = blockIdx.x & 7;         // waves in a block share kpart
    const int rowgroup = blockIdx.x >> 3;        // 0 .. 127
    const int rowbase  = (rowgroup * 4 + wave) * 16;
    const int kstart   = kpart * KCHUNK;

    const int  m     = lane & 15;
    const bool hi    = lane >= 16;
    const int  koffA = hi ? 8  : 0;   // ISA 16-bit A layout: lanes>=16 hold K 8..15,24..31
    const int  koffB = hi ? 8  : 0;   // ISA 16-bit B layout: lanes>=16 hold K 8..15,24..31

    const float* arow = A + (size_t)(rowbase + m) * NROWS + kstart + koffA;

    const unsigned lds_base  = (unsigned)(uintptr_t)&Bs[0][0][0];
    const unsigned ROWBYTES  = BROW * 2;          // 272
    const unsigned BUFBYTES  = 64u * ROWBYTES;    // per-buffer size

    // per-thread staging slots: e = q*128 + tid; row n = e>>4; 16B slot (e&15)
    const int st_n  = 0;  (void)st_n;

    // ---- issue async stage 0 ----
#pragma unroll
    for (int q = 0; q < 8; ++q) {
        int e  = q * 128 + tid;
        int n  = e >> 4;
        int bo = (e & 15) * 16;
        async_copy16(lds_base + (unsigned)n * ROWBYTES + bo,
                     (const char*)(xT + (size_t)n * NROWS + kstart) + bo);
    }

    v8f cacc[4] = {v8f{}, v8f{}, v8f{}, v8f{}};
    ABuf ar = load_a_raw(arow);                   // A pipeline: holds chunk kk

    for (int i = 0; i < NSTAGE; ++i) {
        if (i + 1 < NSTAGE) {
            // issue next stage into the other buffer, then drain current stage
            const int ks_n = (i + 1) * CH;
            const unsigned bb = ((i + 1) & 1) ? BUFBYTES : 0u;
#pragma unroll
            for (int q = 0; q < 8; ++q) {
                int e  = q * 128 + tid;
                int n  = e >> 4;
                int bo = (e & 15) * 16;
                async_copy16(lds_base + bb + (unsigned)n * ROWBYTES + bo,
                             (const char*)(xT + (size_t)n * NROWS + kstart + ks_n) + bo);
            }
            asm volatile("s_wait_asynccnt 0x8" ::: "memory");
        } else {
            asm volatile("s_wait_asynccnt 0x0" ::: "memory");
        }
        __syncthreads();   // stage i fully in LDS for all waves

        const int buf = i & 1;
#pragma unroll
        for (int c = 0; c < CH / 32; ++c) {
            const int kk = i * CH + c * 32;
            // keep the A HBM stream pipelined one chunk ahead
            const int knext = (kk + 32 < KCHUNK) ? (kk + 32) : kk;
            ABuf ar_n = load_a_raw(arow + knext);
            __builtin_prefetch(arow + kk + 512, 0, 3);

            v16h av = cvt_a(ar);
            const int bcol = c * 32 + koffB;
            cacc[0] = WMMA_F16(av, load_b16(&Bs[buf][m][bcol]),      cacc[0]);
            cacc[1] = WMMA_F16(av, load_b16(&Bs[buf][16 + m][bcol]), cacc[1]);
            cacc[2] = WMMA_F16(av, load_b16(&Bs[buf][32 + m][bcol]), cacc[2]);
            cacc[3] = WMMA_F16(av, load_b16(&Bs[buf][48 + m][bcol]), cacc[3]);

            ar = ar_n;
        }
        __syncthreads();   // all waves done reading buf before it is re-staged
    }

    // Partial diag: prod[r] = sum_n x[row, n] * Ax_part[row, n]
    // C/D layout: VGPR r holds (M = r + (hi?8:0), N = (lane&15) + 16*t)
    const int rowoff = hi ? 8 : 0;
    float prod[8];
#pragma unroll
    for (int r = 0; r < 8; ++r) prod[r] = 0.0f;

#pragma unroll
    for (int t = 0; t < 4; ++t) {
        v8h xe = *(const v8h*)(xT + (size_t)(t * 16 + m) * NROWS + rowbase + rowoff);
#pragma unroll
        for (int r = 0; r < 8; ++r) prod[r] += (float)xe[r] * cacc[t][r];
    }

    // Reduce across the 16 lanes of each half-wave (xor net stays within half)
#pragma unroll
    for (int sh = 1; sh < 16; sh <<= 1) {
#pragma unroll
        for (int r = 0; r < 8; ++r) prod[r] += __shfl_xor(prod[r], sh, 32);
    }

    if (m == 0) {   // lanes 0 and 16: rows rowbase..+7 / rowbase+8..+15
#pragma unroll
        for (int r = 0; r < 8; ++r)
            atomicAdd(s + rowbase + rowoff + r, 0.01f * prod[r]);
    }
}

// ---------------------------------------------------------------------------
// Kernel 3: out = F - B*x - s[row]  (vectorized elementwise)
// ---------------------------------------------------------------------------
__global__ void finalize_kernel(const float* __restrict__ x,
                                const float* __restrict__ s,
                                float* __restrict__ out) {
    int idx  = blockIdx.x * blockDim.x + threadIdx.x;
    int base = idx * 4;
    int i = base >> 6;
    v4f xv = *(const v4f*)(x + base);
    float si = s[i];
    v4f o;
#pragma unroll
    for (int j = 0; j < 4; ++j) o[j] = 1.0f - 0.1f * xv[j] - si;
    *(v4f*)(out + base) = o;
}

// ---------------------------------------------------------------------------
extern "C" void kernel_launch(void* const* d_in, const int* in_sizes, int n_in,
                              void* d_out, int out_size, void* d_ws, size_t ws_size,
                              hipStream_t stream) {
    const float* x = (const float*)d_in[1];   // [8192, 64] fp32
    const float* A = (const float*)d_in[2];   // [8192, 8192] fp32
    float* out = (float*)d_out;               // [8192, 64] fp32

    _Float16* xT = (_Float16*)d_ws;                                     // 1 MB
    float*    s  = (float*)((char*)d_ws + (size_t)DIM * NROWS * 2);     // 32 KB

    // 1) cast/transpose x, zero s
    prep_kernel<<<(NROWS * DIM) / 256, 256, 0, stream>>>(x, xT, s);

    // 2) WMMA GEMM + diag partials: 128 rowgroups x 8 K-parts = 1024 blocks
    gemm_diag_kernel<<<128 * KSPLIT, 128, 0, stream>>>(A, xT, s);

    // 3) elementwise finalize
    finalize_kernel<<<(NROWS * DIM / 4) / 256, 256, 0, stream>>>(x, s, out);
}
